// SparseKnnGraph_79542794322568
// MI455X (gfx1250) — matrix operand order repaired and checked by reading, not compiled, not run on hardware
//
#include <hip/hip_runtime.h>
#include <cstdint>
#include <cstddef>

// ---------------------------------------------------------------------------
// SparseKnnGraph for MI455X (gfx1250, wave32, WMMA).
//   B=4, C=192, N=4096, K=9.
//   Phase 1: normalize points, emit f16 hi/lo split + fp32 per-point sq.
//   Phase 2: Gram matrix via v_wmma_f32_16x16x32_f16 (3 WMMAs/chunk for
//            fp32-class accuracy), fused per-row top-10 selection with a
//            branch-skipped quick-max filter. B tile staged in LDS through
//            batched b128 global loads -> b128 LDS stores.
//   Phase 3: merge the two column-split candidate lists, emit edge_index.
// ---------------------------------------------------------------------------

#define BDIM 256
constexpr int BB   = 4;
constexpr int CCH  = 192;     // channels (K-dim of the Gram GEMM)
constexpr int NN   = 4096;    // points per batch
constexpr int KNN  = 9;       // neighbors emitted
constexpr int CP   = 200;     // padded LDS row stride in halves (16B aligned, odd in 16B units)
constexpr int NT   = 64;      // B-tile columns resident in LDS per iteration
constexpr int MBLK = 128;     // rows per workgroup (8 waves * 16)
constexpr int NSPLIT = 2;     // column splits (for occupancy)
constexpr int NCOLS  = NN / NSPLIT;
constexpr int ROWV4  = CCH / 8;              // 24 uint4 per point row
constexpr int CPYN   = NT * ROWV4 / BDIM;    // 6 uint4 copies per thread

typedef __attribute__((ext_vector_type(16))) _Float16 v16h;
typedef __attribute__((ext_vector_type(8)))  _Float16 v8h;
typedef __attribute__((ext_vector_type(8)))  float    v8f;

static __device__ inline v16h cat8(v8h a, v8h b) {
  v16h r;
#pragma unroll
  for (int i = 0; i < 8; ++i) { r[i] = a[i]; r[i + 8] = b[i]; }
  return r;
}

static __device__ inline float waveSum(float v) {
#pragma unroll
  for (int m = 16; m >= 1; m >>= 1) v += __shfl_xor(v, m, 32);
  return v;
}

// --------------------------- Phase 1: normalize ----------------------------
// One wave per point: 6 channels/lane. Reads x[b][c][n] (stride-N across c),
// writes row-major hi/lo f16 (coalesced 64B bursts per lane-step).
__global__ __launch_bounds__(BDIM) void prep_kernel(
    const float* __restrict__ x,
    _Float16* __restrict__ hiG, _Float16* __restrict__ loG,
    float* __restrict__ sqG)
{
  const int wave = threadIdx.x >> 5, lane = threadIdx.x & 31;
  const int p = blockIdx.x * 8 + wave;            // global point id
  const int b = p >> 12, n = p & (NN - 1);
  const float* xb = x + (size_t)b * CCH * NN + n;

  float vals[6];
  float ssq = 0.f;
#pragma unroll
  for (int t = 0; t < 6; ++t) {
    float v = xb[(size_t)(lane + 32 * t) * NN];
    vals[t] = v;
    ssq += v * v;
  }
  ssq = waveSum(ssq);
  const float inv = 1.f / fmaxf(sqrtf(ssq), 1e-12f);

  const size_t rb = (size_t)p * CCH;
  float s2 = 0.f;
#pragma unroll
  for (int t = 0; t < 6; ++t) {
    float v = vals[t] * inv;
    _Float16 h = (_Float16)v;
    _Float16 l = (_Float16)(v - (float)h);
    hiG[rb + lane + 32 * t] = h;
    loG[rb + lane + 32 * t] = l;
    s2 += v * v;
  }
  s2 = waveSum(s2);
  if (lane == 0) sqG[p] = s2;
}

// ---------------------- Phase 2: WMMA Gram + fused top-10 ------------------
__global__ __launch_bounds__(BDIM) void gram_topk_kernel(
    const _Float16* __restrict__ hiG, const _Float16* __restrict__ loG,
    const float* __restrict__ sqG,
    float* __restrict__ lv, int* __restrict__ li)
{
  __shared__ _Float16 Bhi[NT * CP];
  __shared__ _Float16 Blo[NT * CP];
  __shared__ float    Bsq[NT];
  __shared__ float    scr[8][16][20];   // per-wave 16x16 tile; 20-float stride:
                                        // rows 16B-aligned, odd in 16B units.

  const int split = blockIdx.x, mb = blockIdx.y, b = blockIdx.z;
  const int tid = threadIdx.x, wave = tid >> 5, lane = tid & 31;
  const int r15 = lane & 15, hv = lane >> 4;
  const int row = mb * MBLK + wave * 16 + r15;     // row within batch

  // ---- A fragments for this wave's 16 rows, all 6 K-chunks, hi+lo ----
  // 16-bit A 16x32 layout (matches B): lane<16 holds K k0..k0+15,
  // lane>=16 holds K k0+16..k0+31  (hv*16 shift).
  const _Float16* rH = hiG + (size_t)(b * NN + row) * CCH;
  const _Float16* rL = loG + (size_t)(b * NN + row) * CCH;
  v16h ahi[6], alo[6];
#pragma unroll
  for (int k = 0; k < 6; ++k) {
    const int k0 = k * 32 + hv * 16;
    ahi[k] = cat8(*(const v8h*)(rH + k0), *(const v8h*)(rH + k0 + 8));
    alo[k] = cat8(*(const v8h*)(rL + k0), *(const v8h*)(rL + k0 + 8));
  }

  // Register-resident sorted top-10 (value desc) for lanes 0..15. Candidates
  // arrive in increasing index order, so strict '>' vs the 10th entry is an
  // exact implementation of jax.lax.top_k tie-breaking within this split.
  float topv[10]; int topi[10];
#pragma unroll
  for (int j = 0; j < 10; ++j) { topv[j] = -3.4e38f; topi[j] = 0x7fffffff; }
  float thr = -3.4e38f;   // == topv[9]

  const uint4* ghi4 = (const uint4*)hiG;
  const uint4* glo4 = (const uint4*)loG;

  // Per-thread copy slots (fixed: NT*ROWV4 == CPYN*BDIM).
  int cpp[CPYN], cj[CPYN];
#pragma unroll
  for (int u = 0; u < CPYN; ++u) {
    const int i = tid + u * BDIM;
    cpp[u] = i / ROWV4;
    cj[u]  = i % ROWV4;
  }

  for (int it = 0; it < NCOLS / NT; ++it) {
    const int nbase = split * NCOLS + it * NT;
    __syncthreads();   // previous tile's B reads complete

    // Batched B-tile fill: issue all 2*CPYN b128 loads, then b128 LDS stores.
    uint4 th[CPYN], tl[CPYN];
#pragma unroll
    for (int u = 0; u < CPYN; ++u) {
      const size_t g = (size_t)(b * NN + nbase + cpp[u]) * ROWV4 + cj[u];
      th[u] = ghi4[g];
      tl[u] = glo4[g];
    }
#pragma unroll
    for (int u = 0; u < CPYN; ++u) {
      *(uint4*)&Bhi[cpp[u] * CP + 8 * cj[u]] = th[u];
      *(uint4*)&Blo[cpp[u] * CP + 8 * cj[u]] = tl[u];
    }
    if (tid < NT) Bsq[tid] = sqG[b * NN + nbase + tid];
    __syncthreads();

#pragma unroll
    for (int st = 0; st < NT / 16; ++st) {
      // 16-bit B 32x16, this code's contiguous K order (same as A): lane<16 = K 0..15 of column r15,
      // lane>=16 = K 16..31 of column r15 -> one contiguous 32B run.
      v8f acc = {};
      const _Float16* bh = &Bhi[(st * 16 + r15) * CP + hv * 16];
      const _Float16* bl = &Blo[(st * 16 + r15) * CP + hv * 16];
#pragma unroll
      for (int k = 0; k < 6; ++k) {
        v16h bhik = *(const v16h*)(bh + k * 32);
        v16h blok = *(const v16h*)(bl + k * 32);
        // dot ~= lo*hi + hi*lo + hi*hi  (fp32-class accuracy)
        acc = __builtin_amdgcn_wmma_f32_16x16x32_f16(false, alo[k], false, bhik,
                                                     (short)0, acc, false, false);
        acc = __builtin_amdgcn_wmma_f32_16x16x32_f16(false, ahi[k], false, blok,
                                                     (short)0, acc, false, false);
        acc = __builtin_amdgcn_wmma_f32_16x16x32_f16(false, ahi[k], false, bhik,
                                                     (short)0, acc, false, false);
      }
      // score = -dist + const(row) = 2*dot - sq[col]; C/D layout: VGPR j ->
      // M = hv*8 + j, N = r15. Transpose through padded LDS scratch.
      const float bsq = Bsq[st * 16 + r15];
#pragma unroll
      for (int j = 0; j < 8; ++j)
        scr[wave][hv * 8 + j][r15] = 2.f * acc[j] - bsq;

      if (lane < 16) {               // lane L owns row L of this wave's tile
        // Quick filter: row max via 4x b128 LDS loads + v_max tree.
        const float4* rp = (const float4*)(&scr[wave][lane][0]);
        const float4 q0 = rp[0], q1 = rp[1], q2 = rp[2], q3 = rp[3];
        const float m01 = fmaxf(fmaxf(q0.x, q0.y), fmaxf(q0.z, q0.w));
        const float m23 = fmaxf(fmaxf(q1.x, q1.y), fmaxf(q1.z, q1.w));
        const float m45 = fmaxf(fmaxf(q2.x, q2.y), fmaxf(q2.z, q2.w));
        const float m67 = fmaxf(fmaxf(q3.x, q3.y), fmaxf(q3.z, q3.w));
        const float mx = fmaxf(fmaxf(m01, m23), fmaxf(m45, m67));
        if (__builtin_expect(mx > thr, 0)) {
          const int gibase = nbase + st * 16;
#pragma unroll 1
          for (int cc = 0; cc < 16; ++cc) {
            const float v = scr[wave][lane][cc];
            if (v > thr) {
              topv[9] = v; topi[9] = gibase + cc;
#pragma unroll
              for (int j = 9; j > 0; --j) {
                if (topv[j] > topv[j - 1]) {
                  float tv = topv[j]; topv[j] = topv[j - 1]; topv[j - 1] = tv;
                  int   ti = topi[j]; topi[j] = topi[j - 1]; topi[j - 1] = ti;
                }
              }
              thr = topv[9];
            }
          }
        }
      }
    }
  }

  if (lane < 16) {
    const size_t base = (((size_t)(b * NN + row)) * NSPLIT + split) * 10;
#pragma unroll
    for (int j = 0; j < 10; ++j) { lv[base + j] = topv[j]; li[base + j] = topi[j]; }
  }
}

// ------------------- Phase 3: merge splits, emit edges ---------------------
__global__ __launch_bounds__(BDIM) void emit_kernel(
    const float* __restrict__ lv, const int* __restrict__ li,
    int* __restrict__ out)
{
  const int t = blockIdx.x * BDIM + threadIdx.x;
  if (t >= BB * NN) return;
  const int b = t >> 12, n = t & (NN - 1);

  float topv[10]; int topi[10];
#pragma unroll
  for (int j = 0; j < 10; ++j) { topv[j] = -3.4e38f; topi[j] = 0x7fffffff; }

  const size_t base = (size_t)t * (NSPLIT * 10);
  for (int c = 0; c < NSPLIT * 10; ++c) {
    const float v = lv[base + c];
    const int  gi = li[base + c];
    // Cross-split ties must be resolved by index explicitly.
    if (v > topv[9] || (v == topv[9] && gi < topi[9])) {
      topv[9] = v; topi[9] = gi;
#pragma unroll
      for (int j = 9; j > 0; --j) {
        const bool sw = (topv[j] > topv[j - 1]) ||
                        (topv[j] == topv[j - 1] && topi[j] < topi[j - 1]);
        if (sw) {
          float tv = topv[j]; topv[j] = topv[j - 1]; topv[j - 1] = tv;
          int   ti = topi[j]; topi[j] = topi[j - 1]; topi[j - 1] = ti;
        }
      }
    }
  }

  // Drop list[0] (the self/nearest entry), emit entries 1..9.
  const int E = BB * NN * KNN;
#pragma unroll
  for (int k = 1; k <= KNN; ++k) {
    const int e = t * KNN + (k - 1);
    out[e]     = topi[k] + b * NN;   // nn_idx + batch offset
    out[E + e] = n + b * NN;         // center_idx + batch offset
  }
}

// ---------------------------------------------------------------------------
extern "C" void kernel_launch(void* const* d_in, const int* in_sizes, int n_in,
                              void* d_out, int out_size, void* d_ws, size_t ws_size,
                              hipStream_t stream) {
  (void)in_sizes; (void)n_in; (void)out_size; (void)ws_size;
  const float* x = (const float*)d_in[0];

  // Workspace partition (~14.6 MB total).
  char* w = (char*)d_ws;
  _Float16* hiG = (_Float16*)w;
  _Float16* loG = hiG + (size_t)BB * NN * CCH;
  float*    sqG = (float*)(loG + (size_t)BB * NN * CCH);
  float*    lv  = sqG + (size_t)BB * NN;
  int*      li  = (int*)(lv + (size_t)BB * NN * NSPLIT * 10);
  int*      out = (int*)d_out;

  prep_kernel<<<BB * NN / 8, BDIM, 0, stream>>>(x, hiG, loG, sqG);

  dim3 g2(NSPLIT, NN / MBLK, BB);
  gram_topk_kernel<<<g2, BDIM, 0, stream>>>(hiG, loG, sqG, lv, li);

  emit_kernel<<<(BB * NN + BDIM - 1) / BDIM, BDIM, 0, stream>>>(lv, li, out);
}
